// InfoNCELoss_57440892617419
// MI455X (gfx1250) — matrix operand order repaired and checked by reading, not compiled, not run on hardware
//
#include <hip/hip_runtime.h>
#include <hip/hip_bf16.h>

typedef __attribute__((ext_vector_type(16))) _Float16 v16h;
typedef __attribute__((ext_vector_type(8)))  _Float16 v8h;
typedef __attribute__((ext_vector_type(8)))  float    v8f;
typedef __attribute__((ext_vector_type(4)))  float    v4f;

#define B_ROWS 4096
#define P_ROWS 4096
#define N_ROWS 8192
#define DDIM   256
#define POS_TILES (P_ROWS / 16)            // 256
#define TOT_TILES ((P_ROWS + N_ROWS) / 16) // 768
#define CSLICES 16
#define WAVES_PER_WG 4
#define SLICE_STRIDE (CSLICES * WAVES_PER_WG) // 64
#define INV_T_LOG2E 14.4269504088896341f      // (1/0.1) * log2(e)

// ---------------------------------------------------------------------------
// Kernel 1: row-normalize (x / max(||x||, 1e-8)) and split each f32 into an
// f16 hi/lo pair so the WMMA f16 path reaches ~f32 accuracy.
// One wave per row; lane handles 8 contiguous elements.
// ---------------------------------------------------------------------------
__global__ __launch_bounds__(256) void normalize_split(
    const float* __restrict__ x, _Float16* __restrict__ hi,
    _Float16* __restrict__ lo, int rows) {
  const int wave = threadIdx.x >> 5;
  const int lane = threadIdx.x & 31;
  const int row = blockIdx.x * 8 + wave;
  if (row >= rows) return;

  const float* p = x + (size_t)row * DDIM + lane * 8;
  v4f x0 = *(const v4f*)p;
  v4f x1 = *(const v4f*)(p + 4);

  float s = 0.f;
  s = fmaf(x0.x, x0.x, s); s = fmaf(x0.y, x0.y, s);
  s = fmaf(x0.z, x0.z, s); s = fmaf(x0.w, x0.w, s);
  s = fmaf(x1.x, x1.x, s); s = fmaf(x1.y, x1.y, s);
  s = fmaf(x1.z, x1.z, s); s = fmaf(x1.w, x1.w, s);
#pragma unroll
  for (int m = 1; m < 32; m <<= 1) s += __shfl_xor(s, m, 32);

  const float scale = 1.0f / fmaxf(sqrtf(s), 1e-8f);

  float v[8] = {x0.x, x0.y, x0.z, x0.w, x1.x, x1.y, x1.z, x1.w};
  v8h h, l;
#pragma unroll
  for (int i = 0; i < 8; ++i) {
    float y = v[i] * scale;
    _Float16 hh = (_Float16)y;
    h[i] = hh;
    l[i] = (_Float16)(y - (float)hh);
  }
  *(v8h*)(hi + (size_t)row * DDIM + lane * 8) = h;
  *(v8h*)(lo + (size_t)row * DDIM + lane * 8) = l;
}

// ---------------------------------------------------------------------------
// Kernel 2: zero per-row accumulators (atomicAdd targets, cleared every call)
// ---------------------------------------------------------------------------
__global__ void zero_acc(float* __restrict__ p, int n) {
  int i = blockIdx.x * blockDim.x + threadIdx.x;
  if (i < n) p[i] = 0.0f;
}

// ---------------------------------------------------------------------------
// Fragment loaders (wave32 WMMA VGPR layouts, 16-bit data, contiguous k = 16h+s, same for A and B).
// A (16x32, MxK): lane%16 -> M row; low half-wave K 0..15,
//                 high half-wave K 16..31   -> two 16B chunks.
// B (32x16, KxN): lane%16 -> N col (row of row-major P/N matrix);
//                 low half-wave K 0..15, high K 16..31 -> one 32B chunk.
// ---------------------------------------------------------------------------
__device__ __forceinline__ v16h loadA(const _Float16* __restrict__ base,
                                      int row, int kc, int hiG) {
  const _Float16* p = base + ((size_t)row << 8) + kc + (hiG << 4);
  v8h a0 = *(const v8h*)p;
  v8h a1 = *(const v8h*)(p + 8);
  return __builtin_shufflevector(a0, a1, 0, 1, 2, 3, 4, 5, 6, 7, 8, 9, 10, 11,
                                 12, 13, 14, 15);
}

// Per-tile base pointers for the B operand (pos vs neg select folded once,
// lane's column row and half-wave K offset folded in).
struct TilePtrs {
  const _Float16* h;
  const _Float16* l;
};

__device__ __forceinline__ TilePtrs tile_ptrs(
    int t, int l16, int hoff, const _Float16* __restrict__ Phi,
    const _Float16* __restrict__ Plo, const _Float16* __restrict__ Nhi,
    const _Float16* __restrict__ Nlo) {
  const bool isPos = (t < POS_TILES);
  const int col = (isPos ? (t << 4) : ((t - POS_TILES) << 4)) + l16;
  TilePtrs r;
  r.h = (isPos ? Phi : Nhi) + ((size_t)col << 8) + hoff;
  r.l = (isPos ? Plo : Nlo) + ((size_t)col << 8) + hoff;
  return r;
}

// ---------------------------------------------------------------------------
// Kernel 3: fused GEMM + exp + row reductions.
// Grid: (B/16 row blocks, CSLICES).  Block: 128 threads = 4 waves.
// Each wave owns the 16-row anchor block and a strided subset of the 768
// 16-column tiles (first 256 = positives, rest = negatives).
// A hi/lo fragments (all 8 k-chunks) live in registers for the whole kernel;
// B fragments are double-buffered across k-chunks AND across tiles so loads
// overlap wmma and the exp epilogue (no loadcnt-0 drain per chunk).
// ---------------------------------------------------------------------------
__global__ __launch_bounds__(128) void infonce_main(
    const _Float16* __restrict__ Ahi, const _Float16* __restrict__ Alo,
    const _Float16* __restrict__ Phi, const _Float16* __restrict__ Plo,
    const _Float16* __restrict__ Nhi, const _Float16* __restrict__ Nlo,
    float* __restrict__ posExp, float* __restrict__ negExp,
    float* __restrict__ posSum) {
  const int lane = threadIdx.x & 31;
  const int l16 = lane & 15;
  const int hiG = (lane >> 4) & 1;
  const int wave = threadIdx.x >> 5;
  const int rowBase = blockIdx.x * 16;
  const int slice = blockIdx.y * WAVES_PER_WG + wave; // 0..63
  const int hoff = hiG << 4;

  // Preload the full A panel (16 rows x 256 K) hi+lo into registers.
  const int aRow = rowBase + l16;
  v16h aH[8], aL[8];
#pragma unroll
  for (int chunk = 0; chunk < 8; ++chunk) {
    aH[chunk] = loadA(Ahi, aRow, chunk * 32, hiG);
    aL[chunk] = loadA(Alo, aRow, chunk * 32, hiG);
  }

  float accPE[8], accPS[8], accNE[8];
#pragma unroll
  for (int v = 0; v < 8; ++v) { accPE[v] = 0.f; accPS[v] = 0.f; accNE[v] = 0.f; }

  // Pipeline prologue: first tile's first-chunk B fragments in flight.
  TilePtrs cur = tile_ptrs(slice, l16, hoff, Phi, Plo, Nhi, Nlo);
  v16h bh = *(const v16h*)cur.h;
  v16h bl = *(const v16h*)cur.l;

  for (int t = slice; t < TOT_TILES; t += SLICE_STRIDE) {
    const bool isPos = (t < POS_TILES);
    const int tn = t + SLICE_STRIDE;
    // Next tile's pointers (wrap to a valid tile when past the end; the
    // prefetched data is simply discarded after the last iteration).
    const TilePtrs nxt =
        tile_ptrs(tn < TOT_TILES ? tn : slice, l16, hoff, Phi, Plo, Nhi, Nlo);

    v8f c = {};
#pragma unroll
    for (int chunk = 0; chunk < 8; ++chunk) {
      const v16h curbh = bh;
      const v16h curbl = bl;
      // Issue the next chunk's (or next tile's) B loads before consuming.
      const _Float16* nh =
          (chunk < 7) ? (cur.h + (chunk + 1) * 32) : nxt.h;
      const _Float16* nl =
          (chunk < 7) ? (cur.l + (chunk + 1) * 32) : nxt.l;
      bh = *(const v16h*)nh;
      bl = *(const v16h*)nl;
      // c += ah*bh + al*bh + ah*bl   (hi/lo split ~ f32 accuracy)
      c = __builtin_amdgcn_wmma_f32_16x16x32_f16(false, aH[chunk], false,
                                                 curbh, (short)0, c, false,
                                                 false);
      c = __builtin_amdgcn_wmma_f32_16x16x32_f16(false, aL[chunk], false,
                                                 curbh, (short)0, c, false,
                                                 false);
      c = __builtin_amdgcn_wmma_f32_16x16x32_f16(false, aH[chunk], false,
                                                 curbl, (short)0, c, false,
                                                 false);
    }
    cur = nxt;

    // Epilogue overlaps with the next tile's in-flight B loads.
    if (isPos) {
#pragma unroll
      for (int v = 0; v < 8; ++v) {
        float s = c[v];
        accPE[v] += exp2f(s * INV_T_LOG2E);
        accPS[v] += s;
      }
    } else {
#pragma unroll
      for (int v = 0; v < 8; ++v) accNE[v] += exp2f(c[v] * INV_T_LOG2E);
    }
  }

  // C layout: VGPR v, lanes 0-15 -> row (rowBase+v), lanes 16-31 -> row+8.
  // Reduce each accumulator across its 16-lane group, then atomicAdd.
#pragma unroll
  for (int v = 0; v < 8; ++v) {
    float pe = accPE[v], ps = accPS[v], ne = accNE[v];
#pragma unroll
    for (int m = 1; m < 16; m <<= 1) {
      pe += __shfl_xor(pe, m, 32);
      ps += __shfl_xor(ps, m, 32);
      ne += __shfl_xor(ne, m, 32);
    }
    if (l16 == 0) {
      const int row = rowBase + v + (hiG << 3);
      atomicAdd(&posExp[row], pe);
      atomicAdd(&posSum[row], ps);
      atomicAdd(&negExp[row], ne);
    }
  }
}

// ---------------------------------------------------------------------------
// Kernel 4: finalize scalar loss.
// loss_b = log(pos+neg+1e-8) - log(pos);  cov_b = 1 - posSum_b / P
// out = mean(loss) + 0.5*mean(cov)
// ---------------------------------------------------------------------------
__global__ __launch_bounds__(256) void finalize(
    const float* __restrict__ posExp, const float* __restrict__ negExp,
    const float* __restrict__ posSum, float* __restrict__ out) {
  __shared__ float sL[8], sC[8];
  float ls = 0.f, cs = 0.f;
  for (int b = threadIdx.x; b < B_ROWS; b += 256) {
    float pe = posExp[b];
    float all = pe + negExp[b] + 1e-8f;
    ls += logf(all) - logf(pe);
    cs += 1.0f - posSum[b] * (1.0f / (float)P_ROWS);
  }
#pragma unroll
  for (int m = 1; m < 32; m <<= 1) {
    ls += __shfl_xor(ls, m, 32);
    cs += __shfl_xor(cs, m, 32);
  }
  const int wave = threadIdx.x >> 5;
  if ((threadIdx.x & 31) == 0) { sL[wave] = ls; sC[wave] = cs; }
  __syncthreads();
  if (threadIdx.x == 0) {
    float L = 0.f, C = 0.f;
#pragma unroll
    for (int i = 0; i < 8; ++i) { L += sL[i]; C += sC[i]; }
    out[0] = L * (1.0f / (float)B_ROWS) +
             0.5f * C * (1.0f / (float)B_ROWS);
  }
}

// ---------------------------------------------------------------------------
extern "C" void kernel_launch(void* const* d_in, const int* in_sizes, int n_in,
                              void* d_out, int out_size, void* d_ws,
                              size_t ws_size, hipStream_t stream) {
  const float* anchor = (const float*)d_in[0]; // [4096, 256]
  const float* pos    = (const float*)d_in[1]; // [4096, 256]
  const float* neg    = (const float*)d_in[2]; // [8192, 256]
  float* out = (float*)d_out;

  // Workspace layout (16 MB of f16 + 48 KB of f32 accumulators)
  _Float16* Ahi = (_Float16*)d_ws;
  _Float16* Alo = Ahi + (size_t)B_ROWS * DDIM;
  _Float16* Phi = Alo + (size_t)B_ROWS * DDIM;
  _Float16* Plo = Phi + (size_t)P_ROWS * DDIM;
  _Float16* Nhi = Plo + (size_t)P_ROWS * DDIM;
  _Float16* Nlo = Nhi + (size_t)N_ROWS * DDIM;
  float* posExp = (float*)(Nlo + (size_t)N_ROWS * DDIM);
  float* negExp = posExp + B_ROWS;
  float* posSum = negExp + B_ROWS;

  // 1) clear accumulators (atomicAdd targets)
  zero_acc<<<(3 * B_ROWS + 255) / 256, 256, 0, stream>>>(posExp, 3 * B_ROWS);

  // 2) normalize + hi/lo split (one wave per row, 8 rows per block)
  normalize_split<<<B_ROWS / 8, 256, 0, stream>>>(anchor, Ahi, Alo, B_ROWS);
  normalize_split<<<P_ROWS / 8, 256, 0, stream>>>(pos, Phi, Plo, P_ROWS);
  normalize_split<<<N_ROWS / 8, 256, 0, stream>>>(neg, Nhi, Nlo, N_ROWS);

  // 3) fused WMMA GEMM + exp + row reductions
  dim3 grid(B_ROWS / 16, CSLICES);
  infonce_main<<<grid, 32 * WAVES_PER_WG, 0, stream>>>(
      Ahi, Alo, Phi, Plo, Nhi, Nlo, posExp, negExp, posSum);

  // 4) scalar finalize
  finalize<<<1, 256, 0, stream>>>(posExp, negExp, posSum, out);
}
